// TreeLSTM_274877907230
// MI455X (gfx1250) — matrix operand order repaired and checked by reading, not compiled, not run on hardware
//
#include <hip/hip_runtime.h>
#include <hip/hip_fp16.h>
#include <math.h>

// ---------------------------------------------------------------------------
// TreeLSTM on MI455X (gfx1250): f16 WMMA GEMMs (fp32 accumulate), fused cell
// with native v_tanh_f32, LDS-resident fused tail levels.
// ---------------------------------------------------------------------------

typedef __attribute__((ext_vector_type(16))) _Float16 v16h;
typedef __attribute__((ext_vector_type(8)))  _Float16 v8h;
typedef __attribute__((ext_vector_type(8)))  float    v8f;
typedef __attribute__((ext_vector_type(4)))  float    v4f;

#define HID     128
#define DIN     128
#define NLEAVES 262144
#define FIVEH   640
#define PADH    136   // halfs per LDS h row (pad to spread banks)
#define PADC    132   // floats per LDS c row

__device__ __forceinline__ float fast_tanh(float x) {
#if __has_builtin(__builtin_amdgcn_tanhf)
    return __builtin_amdgcn_tanhf(x);              // v_tanh_f32 (gfx1250 TRANS op)
#else
    x = fminf(fmaxf(x, -15.0f), 15.0f);
    float e = __expf(-2.0f * x);
    return (1.0f - e) * __builtin_amdgcn_rcpf(1.0f + e);
#endif
}
__device__ __forceinline__ float fast_sig(float x) {
    return 0.5f * fast_tanh(0.5f * x) + 0.5f;      // sigmoid via tanh
}

// ---------------------------------------------------------------------------
// Pack fp32 weights [128 x 640] into f16 WMMA-B-ready tiles.
// Tile (nt, kt): nt in [0,40) over N (16 cols), kt in [0,4) over K (32 deep).
// Per-lane layout (ISA 7.12.2, 16-bit B 32x16): col = lane&15,
// K = kt*32 + 8*(lane>>4) + (j&7) + 16*(j>>3), j = 0..15 (table order).
// ---------------------------------------------------------------------------
__global__ void pack_weights_kernel(const float* __restrict__ Wx,
                                    const float* __restrict__ UL,
                                    const float* __restrict__ UR,
                                    _Float16* __restrict__ pW,
                                    _Float16* __restrict__ pL,
                                    _Float16* __restrict__ pR)
{
    const int per = DIN * FIVEH;                 // 81920 elements per weight
    int idx = blockIdx.x * 256 + threadIdx.x;    // [0, 3*81920)
    if (idx >= 3 * per) return;
    int w = idx / per;
    int e = idx - w * per;
    int j    = e & 15;
    int lane = (e >> 4) & 31;
    int tile = e >> 9;                           // = nt*4 + kt
    int kt   = tile & 3;
    int nt   = tile >> 2;
    int col  = nt * 16 + (lane & 15);
    int k    = kt * 32 + 8 * (lane >> 4) + (j & 7) + 16 * (j >> 3);
    const float* src = (w == 0) ? Wx : (w == 1) ? UL : UR;
    _Float16*    dst = (w == 0) ? pW : (w == 1) ? pL : pR;
    dst[e] = (_Float16)src[k * FIVEH + col];
}

// ---------------------------------------------------------------------------
// Leaf stage: gates = X @ W_x + b_x ; c = sig(i)*tanh(u); h = sig(o)*tanh(c)
// Block = 256 threads = 8 waves, owns 16 rows. Wave w owns cols [16w,16w+16)
// of each of the 5 gate chunks -> fully register-fused cell.
// ---------------------------------------------------------------------------
__global__ void __launch_bounds__(256) leaf_kernel(
    const float*    __restrict__ X,     // [NLEAVES, DIN] fp32
    const _Float16* __restrict__ pW,    // packed W_x
    const float*    __restrict__ bx,    // [640]
    _Float16*       __restrict__ h_out, // [NLEAVES, HID] f16
    float*          __restrict__ c_out) // [NLEAVES, HID] fp32
{
    const int lane = threadIdx.x & 31;
    const int wv   = threadIdx.x >> 5;          // 0..7
    const int row_base = blockIdx.x * 16;
    const int m15  = lane & 15;
    const int hsel = lane >> 4;                 // 0 or 1
    const int arow = row_base + m15;
    const int col  = wv * 16 + m15;

    v8f acc[5];
    #pragma unroll
    for (int g = 0; g < 5; ++g) {
        float b = bx[g * HID + col];
        #pragma unroll
        for (int e = 0; e < 8; ++e) acc[g][e] = b;
    }

    #pragma unroll
    for (int kt = 0; kt < 4; ++kt) {
        const int    kbase = kt * 32 + 8 * hsel;
        const float* ap    = X + arow * DIN + kbase;
        v4f x0 = *(const v4f*)(ap);
        v4f x1 = *(const v4f*)(ap + 4);
        v4f x2 = *(const v4f*)(ap + 16);
        v4f x3 = *(const v4f*)(ap + 20);
        v16h a;
        #pragma unroll
        for (int t = 0; t < 4; ++t) {
            a[t]      = (_Float16)x0[t];
            a[4 + t]  = (_Float16)x1[t];
            a[8 + t]  = (_Float16)x2[t];
            a[12 + t] = (_Float16)x3[t];
        }
        #pragma unroll
        for (int g = 0; g < 5; ++g) {
            const int nt = g * 8 + wv;
            const v16h b = *(const v16h*)(pW + ((nt * 4 + kt) * 32 + lane) * 16);
            acc[g] = __builtin_amdgcn_wmma_f32_16x16x32_f16(
                false, a, false, b, (short)0, acc[g], false, false);
        }
    }

    #pragma unroll
    for (int r = 0; r < 8; ++r) {
        const int grow = row_base + r + 8 * hsel;
        float ci = fast_sig(acc[0][r]) * fast_tanh(acc[4][r]);
        float hh = fast_sig(acc[3][r]) * fast_tanh(ci);
        c_out[grow * HID + col] = ci;
        h_out[grow * HID + col] = (_Float16)hh;
    }
}

// ---------------------------------------------------------------------------
// Big reduction level: gates = b_x + h_even @ U_L + h_odd @ U_R, full cell.
// ---------------------------------------------------------------------------
__global__ void __launch_bounds__(256) level_kernel(
    const _Float16* __restrict__ h_in,   // [2m, HID] f16
    const float*    __restrict__ c_in,   // [2m, HID] fp32
    const _Float16* __restrict__ pL,
    const _Float16* __restrict__ pR,
    const float*    __restrict__ bx,
    _Float16*       __restrict__ h_out,  // [m, HID] f16
    float*          __restrict__ c_out,  // [m, HID] fp32
    int m)
{
    const int lane = threadIdx.x & 31;
    const int wv   = threadIdx.x >> 5;
    const int row_base = blockIdx.x * 16;
    const int m15  = lane & 15;
    const int hsel = lane >> 4;
    int ar = row_base + m15;
    if (ar >= m) ar = m - 1;
    const int col = wv * 16 + m15;

    // Prefetch the child c lines needed after the WMMA chain (global_prefetch_b8)
    __builtin_prefetch(c_in + (2 * ar) * HID + col, 0, 0);
    __builtin_prefetch(c_in + (2 * ar + 1) * HID + col, 0, 0);

    v8f acc[5];
    #pragma unroll
    for (int g = 0; g < 5; ++g) {
        float b = bx[g * HID + col];
        #pragma unroll
        for (int e = 0; e < 8; ++e) acc[g][e] = b;
    }

    #pragma unroll
    for (int kt = 0; kt < 4; ++kt) {
        const int kbase = kt * 32 + 8 * hsel;
        const _Float16* pe = h_in + (2 * ar)     * HID + kbase;
        const _Float16* po = h_in + (2 * ar + 1) * HID + kbase;
        v8h e0 = *(const v8h*)(pe);
        v8h e1 = *(const v8h*)(pe + 16);
        v8h o0 = *(const v8h*)(po);
        v8h o1 = *(const v8h*)(po + 16);
        v16h aE, aO;
        #pragma unroll
        for (int t = 0; t < 8; ++t) {
            aE[t] = e0[t]; aE[8 + t] = e1[t];
            aO[t] = o0[t]; aO[8 + t] = o1[t];
        }
        #pragma unroll
        for (int g = 0; g < 5; ++g) {
            const int nt = g * 8 + wv;
            const v16h bl = *(const v16h*)(pL + ((nt * 4 + kt) * 32 + lane) * 16);
            const v16h br = *(const v16h*)(pR + ((nt * 4 + kt) * 32 + lane) * 16);
            acc[g] = __builtin_amdgcn_wmma_f32_16x16x32_f16(
                false, aE, false, bl, (short)0, acc[g], false, false);
            acc[g] = __builtin_amdgcn_wmma_f32_16x16x32_f16(
                false, aO, false, br, (short)0, acc[g], false, false);
        }
    }

    #pragma unroll
    for (int r = 0; r < 8; ++r) {
        const int grow = row_base + r + 8 * hsel;
        const int rc   = (grow < m) ? grow : (m - 1);
        float cL = c_in[(2 * rc)     * HID + col];
        float cR = c_in[(2 * rc + 1) * HID + col];
        float cc = fast_sig(acc[0][r]) * fast_tanh(acc[4][r])
                 + fast_sig(acc[1][r]) * cL
                 + fast_sig(acc[2][r]) * cR;
        float hh = fast_sig(acc[3][r]) * fast_tanh(cc);
        if (grow < m) {
            c_out[grow * HID + col] = cc;
            h_out[grow * HID + col] = (_Float16)hh;
        }
    }
}

// ---------------------------------------------------------------------------
// Fused tail: levels m = 32,16,8,4,2,1 in one workgroup, h/c resident in LDS.
// 512 threads = 16 waves = 2 row-tiles x 8 col-waves. In-place update is safe:
// out-row r reads in-rows 2r,2r+1 (>= r), with read -> barrier -> write phases.
// ---------------------------------------------------------------------------
__global__ void __launch_bounds__(512) tail_kernel(
    const _Float16* __restrict__ h_in,  // [64, HID] f16
    const float*    __restrict__ c_in,  // [64, HID] fp32
    const _Float16* __restrict__ pL,
    const _Float16* __restrict__ pR,
    const float*    __restrict__ bx,
    float* __restrict__ out)            // [128] fp32 root h
{
    __shared__ _Float16 sh[64 * PADH];
    __shared__ float    sc[64 * PADC];

    const int tid = threadIdx.x;
    {   // cooperative global -> LDS copy: 64 rows, 8 threads/row, 16 elems each
        const int r    = tid >> 3;
        const int cofs = (tid & 7) * 16;
        v8h a0 = *(const v8h*)(h_in + r * HID + cofs);
        v8h a1 = *(const v8h*)(h_in + r * HID + cofs + 8);
        *(v8h*)(sh + r * PADH + cofs)     = a0;
        *(v8h*)(sh + r * PADH + cofs + 8) = a1;
        v4f c0 = *(const v4f*)(c_in + r * HID + cofs);
        v4f c1 = *(const v4f*)(c_in + r * HID + cofs + 4);
        v4f c2 = *(const v4f*)(c_in + r * HID + cofs + 8);
        v4f c3 = *(const v4f*)(c_in + r * HID + cofs + 12);
        *(v4f*)(sc + r * PADC + cofs)      = c0;
        *(v4f*)(sc + r * PADC + cofs + 4)  = c1;
        *(v4f*)(sc + r * PADC + cofs + 8)  = c2;
        *(v4f*)(sc + r * PADC + cofs + 12) = c3;
    }
    __syncthreads();

    const int lane = tid & 31;
    const int wave = tid >> 5;      // 0..15
    const int wv   = wave & 7;
    const int rt   = wave >> 3;     // 0..1
    const int m15  = lane & 15;
    const int hsel = lane >> 4;
    const int col  = wv * 16 + m15;
    const int row_base = rt * 16;

    float bias[5];
    #pragma unroll
    for (int g = 0; g < 5; ++g) bias[g] = bx[g * HID + col];

    for (int m = 32; m >= 1; m >>= 1) {
        const bool active = (row_base < m);      // uniform per wave -> EXEC all-1s
        float cc[8], hh[8];
        if (active) {
            v8f acc[5];
            #pragma unroll
            for (int g = 0; g < 5; ++g)
                #pragma unroll
                for (int e = 0; e < 8; ++e) acc[g][e] = bias[g];
            int ar = row_base + m15;
            if (ar >= m) ar = m - 1;
            #pragma unroll
            for (int kt = 0; kt < 4; ++kt) {
                const int kbase = kt * 32 + 8 * hsel;
                const _Float16* pe = sh + (2 * ar)     * PADH + kbase;
                const _Float16* po = sh + (2 * ar + 1) * PADH + kbase;
                v8h e0 = *(const v8h*)(pe);
                v8h e1 = *(const v8h*)(pe + 16);
                v8h o0 = *(const v8h*)(po);
                v8h o1 = *(const v8h*)(po + 16);
                v16h aE, aO;
                #pragma unroll
                for (int t = 0; t < 8; ++t) {
                    aE[t] = e0[t]; aE[8 + t] = e1[t];
                    aO[t] = o0[t]; aO[8 + t] = o1[t];
                }
                #pragma unroll
                for (int g = 0; g < 5; ++g) {
                    const int nt = g * 8 + wv;
                    const v16h bl = *(const v16h*)(pL + ((nt * 4 + kt) * 32 + lane) * 16);
                    const v16h br = *(const v16h*)(pR + ((nt * 4 + kt) * 32 + lane) * 16);
                    acc[g] = __builtin_amdgcn_wmma_f32_16x16x32_f16(
                        false, aE, false, bl, (short)0, acc[g], false, false);
                    acc[g] = __builtin_amdgcn_wmma_f32_16x16x32_f16(
                        false, aO, false, br, (short)0, acc[g], false, false);
                }
            }
            #pragma unroll
            for (int r = 0; r < 8; ++r) {
                const int grow = row_base + r + 8 * hsel;
                const int rc   = (grow < m) ? grow : (m - 1);
                float cL = sc[(2 * rc)     * PADC + col];
                float cR = sc[(2 * rc + 1) * PADC + col];
                cc[r] = fast_sig(acc[0][r]) * fast_tanh(acc[4][r])
                      + fast_sig(acc[1][r]) * cL
                      + fast_sig(acc[2][r]) * cR;
                hh[r] = fast_sig(acc[3][r]) * fast_tanh(cc[r]);
            }
        }
        __syncthreads();                          // all LDS reads of this level done
        if (active) {
            #pragma unroll
            for (int r = 0; r < 8; ++r) {
                const int grow = row_base + r + 8 * hsel;
                if (grow < m) {
                    sc[grow * PADC + col] = cc[r];
                    sh[grow * PADH + col] = (_Float16)hh[r];
                    if (m == 1 && grow == 0) out[col] = hh[r];  // fp32 root
                }
            }
        }
        __syncthreads();                          // writes visible for next level
    }
}

// ---------------------------------------------------------------------------
// Launcher. Workspace layout (bytes):
//   [0)       pW  f16 81920   | [163840) pL | [327680) pR
//   [491520)  h0  f16 262144*128 = 67108864
//   [+)       h1  f16 131072*128 = 33554432
//   [+)       c0  f32 262144*128 = 134217728
//   [+)       c1  f32 131072*128 = 67108864
// ---------------------------------------------------------------------------
extern "C" void kernel_launch(void* const* d_in, const int* in_sizes, int n_in,
                              void* d_out, int out_size, void* d_ws, size_t ws_size,
                              hipStream_t stream)
{
    const float* leaf = (const float*)d_in[0];   // [262144, 128]
    const float* Wx   = (const float*)d_in[1];   // [128, 640]
    const float* bx   = (const float*)d_in[2];   // [640]
    const float* UL   = (const float*)d_in[3];   // [128, 640]
    const float* UR   = (const float*)d_in[4];   // [128, 640]
    float* out = (float*)d_out;                  // [128]

    char* ws = (char*)d_ws;
    _Float16* pW = (_Float16*)(ws);
    _Float16* pL = (_Float16*)(ws + 163840);
    _Float16* pR = (_Float16*)(ws + 327680);
    size_t off = 491520;
    _Float16* h0 = (_Float16*)(ws + off);            off += 67108864ull;
    _Float16* h1 = (_Float16*)(ws + off);            off += 33554432ull;
    float*    c0 = (float*)(ws + off);               off += 134217728ull;
    float*    c1 = (float*)(ws + off);

    // 1) Pack weights to f16 WMMA tiles
    pack_weights_kernel<<<960, 256, 0, stream>>>(Wx, UL, UR, pW, pL, pR);

    // 2) Leaf stage
    leaf_kernel<<<NLEAVES / 16, 256, 0, stream>>>(leaf, pW, bx, h0, c0);

    // 3) Big levels m = 131072 .. 64, ping-ponging buffers
    _Float16* hin = h0;  float* cin = c0;
    _Float16* hout = h1; float* cout = c1;
    int n = NLEAVES;
    while (n > 64) {
        const int m = n >> 1;
        const int blocks = (m + 15) / 16;
        level_kernel<<<blocks, 256, 0, stream>>>(hin, cin, pL, pR, bx,
                                                 hout, cout, m);
        _Float16* th = hin; hin = hout; hout = th;
        float*    tc = cin; cin = cout; cout = tc;
        n = m;
    }

    // 4) Fused tail: levels m = 32..1 in one LDS-resident workgroup
    tail_kernel<<<1, 512, 0, stream>>>(hin, cin, pL, pR, bx, out);
}
